// SubgraphGNN_19490561589477
// MI455X (gfx1250) — matrix operand order repaired and checked by reading, not compiled, not run on hardware
//
#include <hip/hip_runtime.h>
#include <hip/hip_bf16.h>
#include <stdint.h>
#include <stddef.h>

// ---------------------------------------------------------------------------
// SubgraphGNN pipeline for MI455X (gfx1250, wave32, WMMA)
//   - GEMMs via v_wmma_f32_16x16x32_bf16 (f32 accum, bf16 operands)
//   - weight tile staged to LDS via GLOBAL_LOAD_ASYNC_TO_LDS_B128 when the
//     toolchain exposes the builtin (ASYNCcnt-tracked), uint4 copy otherwise
//   - edge scatter: one wave per edge, global_atomic_add_f32 x4 per lane
// ---------------------------------------------------------------------------

#define FDIM 128
#define NGRAPH 64

typedef __attribute__((ext_vector_type(16))) __bf16 bf16x16;
typedef __attribute__((ext_vector_type(8)))  float  f32x8;
typedef __attribute__((ext_vector_type(4)))  int    i32x4;

#if defined(__gfx1250__) && \
    __has_builtin(__builtin_amdgcn_global_load_async_to_lds_b128) && \
    __has_builtin(__builtin_amdgcn_s_wait_asynccnt)
#define USE_ASYNC_LDS 1
typedef __attribute__((address_space(1))) i32x4* gbl_i32x4_ptr;
typedef __attribute__((address_space(3))) i32x4* lds_i32x4_ptr;
#else
#define USE_ASYNC_LDS 0
#endif

// ------------------------------ utility kernels ----------------------------

__global__ __launch_bounds__(256) void k_zero(float* __restrict__ p, int nElem) {
    int i = blockIdx.x * 256 + threadIdx.x;
    if (i < nElem) p[i] = 0.0f;
}

// W [128,128] row-major f32  ->  Wt [n][k] bf16 (i.e. W transposed, column-major)
__global__ __launch_bounds__(256) void k_wt_bf16(const float* __restrict__ W,
                                                 __bf16* __restrict__ Wt) {
    int i = blockIdx.x * 256 + threadIdx.x;      // 0 .. 16383
    int k = i >> 7, n = i & 127;                 // W[k][n]
    Wt[n * FDIM + k] = (__bf16)W[i];
}

__global__ __launch_bounds__(256) void k_degree(const int* __restrict__ ei,
                                                float* __restrict__ deg, int E) {
    int e = blockIdx.x * 256 + threadIdx.x;
    if (e < E) atomicAdd(&deg[ei[E + e]], 1.0f);   // dst = ei[1][e]
}

__global__ __launch_bounds__(256) void k_dinv(float* __restrict__ deg, int n) {
    int i = blockIdx.x * 256 + threadIdx.x;
    if (i < n) deg[i] = rsqrtf(deg[i] + 1.0f);     // deg incl. self loop
}

// ------------------------------ WMMA GEMM ----------------------------------
// XW[n,128] = X[n,128] @ W[128,128]   (W provided transposed, bf16, LDS-staged)
// Block = 256 threads = 8 waves; each wave computes one 16-row x 128-col strip.
__global__ __launch_bounds__(256) void k_gemm_wmma(const float* __restrict__ X,
                                                   const __bf16* __restrict__ Wt,
                                                   float* __restrict__ XW,
                                                   int nrows) {
    __shared__ __align__(32) __bf16 sW[FDIM * FDIM];   // 32 KB, [n][k]

    int tid = threadIdx.x;
    // cooperative LDS fill: 16384 bf16 = 2048 x 16B chunks
#if USE_ASYNC_LDS
    for (int i = tid; i < (FDIM * FDIM) / 8; i += 256) {
        __builtin_amdgcn_global_load_async_to_lds_b128(
            (gbl_i32x4_ptr)(Wt + i * 8),
            (lds_i32x4_ptr)(sW + i * 8),
            /*offset=*/0, /*cpol=*/0);
    }
    __builtin_amdgcn_s_wait_asynccnt(0);
#else
    for (int i = tid; i < (FDIM * FDIM) / 8; i += 256)
        ((uint4*)sW)[i] = ((const uint4*)Wt)[i];
#endif
    __syncthreads();

    int wave  = tid >> 5;
    int lane  = tid & 31;
    int half  = lane >> 4;                 // 0: lanes 0-15, 1: lanes 16-31
    int mtile = blockIdx.x * 8 + wave;
    int row0  = mtile * 16;
    bool fullTile = (row0 + 16) <= nrows;  // true for all but the last tile

    int arow = row0 + (lane & 15);
    if (arow >= nrows) arow = nrows - 1;   // clamped read; stores are guarded
    const float* xrow = X + (size_t)arow * FDIM;

    // A fragments: 16x32 bf16 per k-step, 4 k-steps cover K=128.
    // k order chosen here, same for B: lanes 0-15 hold K = kk*32 + 0..15,
    //                         lanes 16-31 hold K = kk*32 + 16..31.
    bf16x16 afrag[4];
#pragma unroll
    for (int kk = 0; kk < 4; ++kk) {
        int kb = kk * 32 + half * 16;
#pragma unroll
        for (int j = 0; j < 8; ++j) afrag[kk][j]     = (__bf16)xrow[kb + j];
#pragma unroll
        for (int j = 0; j < 8; ++j) afrag[kk][8 + j] = (__bf16)xrow[kb + 8 + j];
    }

    // 8 column tiles of 16; A fragments reused across all of them.
#pragma unroll
    for (int nt = 0; nt < 8; ++nt) {
        f32x8 acc = {};
        int ncol = nt * 16 + (lane & 15);
        // B in the same k order as A: lanes 0-15 hold K = kk*32 + 0..15 of their
        // column, lanes 16-31 hold K = kk*32 + 16..31 -> contiguous in Wt[n][k].
        const __bf16* wcol = sW + (size_t)ncol * FDIM + half * 16;
#pragma unroll
        for (int kk = 0; kk < 4; ++kk) {
            bf16x16 bfrag = *(const bf16x16*)(wcol + kk * 32);
            acc = __builtin_amdgcn_wmma_f32_16x16x32_bf16(
                      false, afrag[kk], false, bfrag, (short)0, acc, false, false);
        }
        // C/D layout: VGPR r, lanes 0-15 -> M=r; lanes 16-31 -> M=r+8.
        // Each store is wave-coalesced: lanes 0-15 cover 16 consecutive
        // columns of one row, lanes 16-31 the same columns of row+8.
        float* obase = XW + (size_t)(row0 + 8 * half) * FDIM + nt * 16 + (lane & 15);
        if (fullTile) {
#pragma unroll
            for (int r = 0; r < 8; ++r)
                obase[(size_t)r * FDIM] = acc[r];
        } else {
#pragma unroll
            for (int r = 0; r < 8; ++r) {
                int rr = row0 + r + 8 * half;
                if (rr < nrows)
                    XW[(size_t)rr * FDIM + nt * 16 + (lane & 15)] = acc[r];
            }
        }
    }
}

// ------------------------------ edge scatter -------------------------------
// One wave32 per edge: lane owns 4 features. msg = xw[src] * dinv[src]*dinv[dst]
__global__ __launch_bounds__(256) void k_scatter(const float* __restrict__ XW,
                                                 const int* __restrict__ ei,
                                                 const float* __restrict__ dinv,
                                                 float* __restrict__ AGG, int E) {
    int gid  = blockIdx.x * 256 + threadIdx.x;
    int e    = gid >> 5;
    int lane = gid & 31;
    if (e >= E) return;
    int s = ei[e], d = ei[E + e];
    float norm = dinv[s] * dinv[d];
    float* ad = AGG + (size_t)d * FDIM + lane * 4;
    __builtin_prefetch(ad, 1, 0);                              // global_prefetch_b8
    float4 v = ((const float4*)(XW + (size_t)s * FDIM))[lane];
    atomicAdd(ad + 0, v.x * norm);
    atomicAdd(ad + 1, v.y * norm);
    atomicAdd(ad + 2, v.z * norm);
    atomicAdd(ad + 3, v.w * norm);
}

// agg = relu(agg + xw * dinv^2 + b)   (self-loop term + bias + relu), in place
__global__ __launch_bounds__(256) void k_epilogue(float* __restrict__ AGG,
                                                  const float* __restrict__ XW,
                                                  const float* __restrict__ dinv,
                                                  const float* __restrict__ bias,
                                                  int n) {
    int i = blockIdx.x * 256 + threadIdx.x;
    if (i >= n * FDIM) return;
    int node = i >> 7, f = i & 127;
    float di = dinv[node];
    float v  = AGG[i] + XW[i] * (di * di) + bias[f];
    AGG[i]   = fmaxf(v, 0.0f);
}

// p = sigmoid(H @ fcw + fcb), one thread per node (K=128 dot)
__global__ __launch_bounds__(256) void k_selector(const float* __restrict__ H,
                                                  const float* __restrict__ fcw,
                                                  const float* __restrict__ fcb,
                                                  float* __restrict__ p, int n) {
    int i = blockIdx.x * 256 + threadIdx.x;
    if (i >= n) return;
    const float* h = H + (size_t)i * FDIM;
    float acc = fcb[0];
#pragma unroll 8
    for (int f = 0; f < FDIM; ++f) acc += h[f] * fcw[f];
    p[i] = 1.0f / (1.0f + expf(-acc));
}

__global__ __launch_bounds__(256) void k_gate(const float* __restrict__ X,
                                              const float* __restrict__ p,
                                              float* __restrict__ XP, int n) {
    int i = blockIdx.x * 256 + threadIdx.x;
    if (i >= n * FDIM) return;
    XP[i] = X[i] * p[i >> 7];
}

__global__ __launch_bounds__(256) void k_pool_accum(const float* __restrict__ Z,
                                                    const int* __restrict__ batch,
                                                    float* __restrict__ sums,
                                                    float* __restrict__ cnts, int n) {
    int i = blockIdx.x * 256 + threadIdx.x;
    if (i >= n * FDIM) return;
    int node = i >> 7, f = i & 127;
    int g = batch[node];
    atomicAdd(&sums[g * FDIM + f], Z[i]);
    if (f == 0) atomicAdd(&cnts[g], 1.0f);
}

__global__ __launch_bounds__(256) void k_pool_div(const float* __restrict__ sums,
                                                  const float* __restrict__ cnts,
                                                  float* __restrict__ out) {
    int i = blockIdx.x * 256 + threadIdx.x;
    if (i >= NGRAPH * FDIM) return;
    out[i] = sums[i] / fmaxf(cnts[i >> 7], 1.0f);
}

// ------------------------------ host driver --------------------------------

static inline char* carve(char*& cur, size_t bytes) {
    char* p = cur;
    cur += (bytes + 255) & ~(size_t)255;
    return p;
}

extern "C" void kernel_launch(void* const* d_in, const int* in_sizes, int n_in,
                              void* d_out, int out_size, void* d_ws, size_t ws_size,
                              hipStream_t stream) {
    const float* x_H    = (const float*)d_in[0];
    const int*   ei_H   = (const int*)d_in[1];
    const float* x_G    = (const float*)d_in[2];
    const int*   ei_G   = (const int*)d_in[3];
    const int*   bat_H  = (const int*)d_in[4];
    const int*   bat_G  = (const int*)d_in[5];
    const float* sel_W1 = (const float*)d_in[6];
    const float* sel_b1 = (const float*)d_in[7];
    const float* sel_W2 = (const float*)d_in[8];
    const float* sel_b2 = (const float*)d_in[9];
    const float* sel_fcw = (const float*)d_in[10];
    const float* sel_fcb = (const float*)d_in[11];
    const float* emb_W1 = (const float*)d_in[12];
    const float* emb_b1 = (const float*)d_in[13];
    const float* emb_W2 = (const float*)d_in[14];
    const float* emb_b2 = (const float*)d_in[15];

    const int n = in_sizes[0] / FDIM;   // 100000
    const int e = in_sizes[1] / 2;      // 1600000

    float* out   = (float*)d_out;
    float* out_hF = out;                       // [64,128]
    float* out_hG = out + NGRAPH * FDIM;       // [64,128]
    float* out_p  = out + 2 * NGRAPH * FDIM;   // [N]

    // workspace carve
    char* cur = (char*)d_ws;
    float*  XW    = (float*)carve(cur, (size_t)n * FDIM * 4);
    float*  bufA  = (float*)carve(cur, (size_t)n * FDIM * 4);
    float*  bufB  = (float*)carve(cur, (size_t)n * FDIM * 4);
    float*  dinvH = (float*)carve(cur, (size_t)n * 4);
    float*  dinvG = (float*)carve(cur, (size_t)n * 4);
    __bf16* WtS1  = (__bf16*)carve(cur, (size_t)FDIM * FDIM * 2);
    __bf16* WtS2  = (__bf16*)carve(cur, (size_t)FDIM * FDIM * 2);
    __bf16* WtE1  = (__bf16*)carve(cur, (size_t)FDIM * FDIM * 2);
    __bf16* WtE2  = (__bf16*)carve(cur, (size_t)FDIM * FDIM * 2);
    float*  sums  = (float*)carve(cur, (size_t)NGRAPH * FDIM * 4);
    float*  cnts  = (float*)carve(cur, (size_t)NGRAPH * 4);

    const int gN   = (n + 255) / 256;
    const int gNF  = (n * FDIM + 255) / 256;
    const int gE   = (e + 255) / 256;
    const int gE32 = (e * 32 + 255) / 256;
    const int gW   = (FDIM * FDIM + 255) / 256;
    const int mt   = (n + 15) / 16;
    const int gG   = (mt + 7) / 8;            // gemm blocks (8 waves per block)

    // degrees -> dinv (in place), per edge set
    k_zero  <<<gN, 256, 0, stream>>>(dinvH, n);
    k_degree<<<gE, 256, 0, stream>>>(ei_H, dinvH, e);
    k_dinv  <<<gN, 256, 0, stream>>>(dinvH, n);
    k_zero  <<<gN, 256, 0, stream>>>(dinvG, n);
    k_degree<<<gE, 256, 0, stream>>>(ei_G, dinvG, e);
    k_dinv  <<<gN, 256, 0, stream>>>(dinvG, n);

    // transpose+convert the four 128x128 weight matrices to bf16
    k_wt_bf16<<<gW, 256, 0, stream>>>(sel_W1, WtS1);
    k_wt_bf16<<<gW, 256, 0, stream>>>(sel_W2, WtS2);
    k_wt_bf16<<<gW, 256, 0, stream>>>(emb_W1, WtE1);
    k_wt_bf16<<<gW, 256, 0, stream>>>(emb_W2, WtE2);

    auto conv = [&](const float* in, const __bf16* Wt, const float* bias,
                    const int* ei, const float* dinv, float* outbuf) {
        k_gemm_wmma<<<gG,   256, 0, stream>>>(in, Wt, XW, n);
        k_zero     <<<gNF,  256, 0, stream>>>(outbuf, n * FDIM);
        k_scatter  <<<gE32, 256, 0, stream>>>(XW, ei, dinv, outbuf, e);
        k_epilogue <<<gNF,  256, 0, stream>>>(outbuf, XW, dinv, bias, n);
    };

    // selector branch on H
    conv(x_H,  WtS1, sel_b1, ei_H, dinvH, bufA);
    conv(bufA, WtS2, sel_b2, ei_H, dinvH, bufB);
    k_selector<<<gN, 256, 0, stream>>>(bufB, sel_fcw, sel_fcb, out_p, n);

    // embed F: gated x_H
    k_gate<<<gNF, 256, 0, stream>>>(x_H, out_p, bufA, n);
    conv(bufA, WtE1, emb_b1, ei_H, dinvH, bufB);
    conv(bufB, WtE2, emb_b2, ei_H, dinvH, bufA);
    k_zero      <<<(NGRAPH * FDIM + 255) / 256, 256, 0, stream>>>(sums, NGRAPH * FDIM);
    k_zero      <<<1, 256, 0, stream>>>(cnts, NGRAPH);
    k_pool_accum<<<gNF, 256, 0, stream>>>(bufA, bat_H, sums, cnts, n);
    k_pool_div  <<<(NGRAPH * FDIM + 255) / 256, 256, 0, stream>>>(sums, cnts, out_hF);

    // embed G
    conv(x_G,  WtE1, emb_b1, ei_G, dinvG, bufB);
    conv(bufB, WtE2, emb_b2, ei_G, dinvG, bufA);
    k_zero      <<<(NGRAPH * FDIM + 255) / 256, 256, 0, stream>>>(sums, NGRAPH * FDIM);
    k_zero      <<<1, 256, 0, stream>>>(cnts, NGRAPH);
    k_pool_accum<<<gNF, 256, 0, stream>>>(bufA, bat_G, sums, cnts, n);
    k_pool_div  <<<(NGRAPH * FDIM + 255) / 256, 256, 0, stream>>>(sums, cnts, out_hG);

    (void)n_in; (void)out_size; (void)ws_size;
}
